// Mamba2_37855841747445
// MI455X (gfx1250) — matrix operand order repaired and checked by reading, not compiled, not run on hardware
//
#include <hip/hip_runtime.h>

typedef __bf16 bf16_t;
typedef __attribute__((ext_vector_type(16))) __bf16 v16bf;
typedef __attribute__((ext_vector_type(8)))  float  v8f;
typedef unsigned int u32;
typedef __attribute__((ext_vector_type(4))) u32 v4u;
typedef __attribute__((ext_vector_type(8))) int  v8i;
typedef __attribute__((ext_vector_type(4))) int  v4i;

#define BATCH   2
#define SEQL    4096
#define DMODEL  768
#define DINNER  1536
#define NH      24
#define HD      64
#define DSTATE  64
#define NC      64
#define CONVCH  1664
#define DPROJ   3224
#define NROWS   (BATCH * SEQL)   // 8192

// ---------- helpers ----------
__device__ inline bf16_t f2bf(float f) {
  union { float f; unsigned u; } a; a.f = f;
  unsigned r = a.u + 0x7FFFu + ((a.u >> 16) & 1u);   // round-to-nearest-even
  union { unsigned short s; bf16_t b; } o; o.s = (unsigned short)(r >> 16);
  return o.b;
}
__device__ inline float softplus_f(float x) {
  return (x > 20.f) ? x : log1pf(__expf(x));
}
__device__ inline float silu_f(float x) { return x / (1.f + __expf(-x)); }
__device__ inline v8f zero8() { v8f z = {0.f,0.f,0.f,0.f,0.f,0.f,0.f,0.f}; return z; }
// WMMA operand swizzles (wave32 bf16 16x16x32; contiguous k = ks*32 + 16*hi + e, A and B alike)
__device__ inline int swzA(int ks, int hi, int e) { return ks*32 + hi*16 + e; }
__device__ inline int swzB(int ks, int hi, int e) { return ks*32 + hi*16 + e; }

// group-segment byte offset of a __shared__ object (flat addr low 32 bits == LDS offset)
__device__ inline unsigned lds_off(const void* p) {
  return (unsigned)(unsigned long long)p;
}

// ---------- Tensor Data Mover: 2-D tile global -> LDS ----------
// D# per CDNA5 ISA: group0 = {count, lds_addr, global_addr, type=2},
// group1 = {flags/pad, tensor_dim0/1, tile_dim0/1, tensor_dim0_stride}.
// tensor_dim* passed as extents remaining from the tile start (OOB reads -> 0).
// pad: interval code 3 = 16 DWORDs (one 32-bf16 row), amount code 3 = 4 DWORDs
// -> LDS row pitch = 40 bf16, matching the WMMA fragment swizzle reads.
__device__ inline void tdm_load_2d(unsigned lds_byte_off, const void* gptr,
                                   u32 tensor_d0, u32 tensor_d1,
                                   u32 tile_d0, u32 tile_d1,
                                   unsigned long long stride0_elems) {
  unsigned long long ga = (unsigned long long)gptr;
  v4u g0;
  g0[0] = 1u;                                            // count = 1 valid descriptor
  g0[1] = lds_byte_off;                                  // lds_addr
  g0[2] = (u32)ga;                                       // global_addr[31:0]
  g0[3] = ((u32)(ga >> 32) & 0x01FFFFFFu) | (2u << 30);  // global_addr[56:32] | type=2
  v8i g1;
  g1[0] = (int)((1u << 16) |                             // data_size = 1 -> 2 bytes
                (1u << 20) |                             // pad_enable
                (3u << 22) |                             // pad_interval: 16 DWORDs
                (3u << 25));                             // pad_amount: 4 DWORDs
  g1[1] = (int)((tensor_d0 & 0xFFFFu) << 16);            // tensor_dim0[15:0]
  g1[2] = (int)((tensor_d0 >> 16) | ((tensor_d1 & 0xFFFFu) << 16));
  g1[3] = (int)((tensor_d1 >> 16) | (tile_d0 << 16));    // tile_dim0
  g1[4] = (int)(tile_d1 & 0xFFFFu);                      // tile_dim1 (tile_dim2 = 0)
  g1[5] = (int)(u32)(stride0_elems & 0xFFFFFFFFull);     // tensor_dim0_stride[31:0]
  g1[6] = (int)(u32)((stride0_elems >> 32) & 0xFFFFull); // stride[47:32] (dim1_stride=0)
  g1[7] = 0;
  v4i z4 = {0, 0, 0, 0};
#if defined(__clang_major__) && (__clang_major__ >= 23)
  v8i z8 = {0, 0, 0, 0, 0, 0, 0, 0};
  __builtin_amdgcn_tensor_load_to_lds(g0, g1, z4, z4, z8, 0);
#else
  __builtin_amdgcn_tensor_load_to_lds(g0, g1, z4, z4, 0);
#endif
}

// ---------- f32 -> bf16 convert ----------
__global__ __launch_bounds__(256) void convert_kernel(const float* __restrict__ src,
                                                      bf16_t* __restrict__ dst, size_t n) {
  size_t i = (size_t)blockIdx.x * 256 + threadIdx.x;
  if (i < n) dst[i] = f2bf(src[i]);
}

// ---------- bf16 GEMM: C[M x N] = A[M x K] * B[N x K]^T, fp32 accumulate ----------
// 128x128 tile / workgroup, 8 waves each 64x32 (4x2 WMMA tiles).
// Global->LDS staging via TDM (wave 0 issues, TENSORcnt-gated), double-buffered.
__global__ __launch_bounds__(256) void gemm_bf16(const bf16_t* __restrict__ A,
                                                 const bf16_t* __restrict__ Bw,
                                                 float* __restrict__ C,
                                                 int M, int N, int K, int ldc) {
  __shared__ bf16_t As[2][128][40];
  __shared__ bf16_t Bs[2][128][40];
  const int tid  = threadIdx.x;
  const int lane = tid & 31;
  const int w    = tid >> 5;          // 8 waves
  const int wmq  = w >> 2;            // 0..1 : 64-row band
  const int wnq  = w & 3;             // 0..3 : 32-col band
  const int m0   = blockIdx.y * 128;
  const int n0   = blockIdx.x * 128;
  const int mrow = lane & 15, hi = lane >> 4;
  const u32 tileN = (u32)((N - n0) < 128 ? (N - n0) : 128);

  v8f acc[4][2];
#pragma unroll
  for (int i = 0; i < 4; ++i)
#pragma unroll
    for (int j = 0; j < 2; ++j) acc[i][j] = zero8();

  const int nsteps = K / 32;
  if (w == 0) {
    tdm_load_2d(lds_off(&As[0][0][0]), A + (size_t)m0 * K,
                (u32)K, (u32)(M - m0), 32u, 128u, (unsigned long long)K);
    tdm_load_2d(lds_off(&Bs[0][0][0]), Bw + (size_t)n0 * K,
                (u32)K, (u32)(N - n0), 32u, tileN, (unsigned long long)K);
  }
  for (int s = 0; s < nsteps; ++s) {
    const int buf = s & 1;
    if (w == 0) {
      if (s + 1 < nsteps) {
        const int k1 = (s + 1) * 32;
        tdm_load_2d(lds_off(&As[buf ^ 1][0][0]), A + (size_t)m0 * K + k1,
                    (u32)(K - k1), (u32)(M - m0), 32u, 128u, (unsigned long long)K);
        tdm_load_2d(lds_off(&Bs[buf ^ 1][0][0]), Bw + (size_t)n0 * K + k1,
                    (u32)(K - k1), (u32)(N - n0), 32u, tileN, (unsigned long long)K);
        __builtin_amdgcn_s_wait_tensorcnt(2);   // current buffer's pair complete
      } else {
        __builtin_amdgcn_s_wait_tensorcnt(0);
      }
    }
    __syncthreads();

    v16bf af[4], bfr[2];
#pragma unroll
    for (int i = 0; i < 4; ++i) {
      int rbase = wmq * 64 + i * 16 + mrow;
#pragma unroll
      for (int e = 0; e < 16; ++e) af[i][e] = As[buf][rbase][swzA(0, hi, e)];
    }
#pragma unroll
    for (int j = 0; j < 2; ++j) {
      int nbase = wnq * 32 + j * 16 + mrow;
#pragma unroll
      for (int e = 0; e < 16; ++e) bfr[j][e] = Bs[buf][nbase][swzB(0, hi, e)];
    }
#pragma unroll
    for (int i = 0; i < 4; ++i)
#pragma unroll
      for (int j = 0; j < 2; ++j)
        acc[i][j] = __builtin_amdgcn_wmma_f32_16x16x32_bf16(
            false, af[i], false, bfr[j], (short)0, acc[i][j], false, false);
    __syncthreads();   // protect buf from next iteration's TDM overwrite (WAR)
  }
  // store
#pragma unroll
  for (int i = 0; i < 4; ++i)
#pragma unroll
    for (int j = 0; j < 2; ++j)
#pragma unroll
      for (int r = 0; r < 8; ++r) {
        int mm = m0 + wmq * 64 + i * 16 + r + hi * 8;
        int nn = n0 + wnq * 32 + j * 16 + mrow;
        if (nn < N) C[(size_t)mm * ldc + nn] = acc[i][j][r];
      }
}

// ---------- depthwise causal conv(4) + SiLU + split ----------
__global__ __launch_bounds__(256) void conv_kernel(const float* __restrict__ zx,
                                                   const float* __restrict__ conv_w,
                                                   const float* __restrict__ conv_b,
                                                   float* __restrict__ xconv,
                                                   float* __restrict__ Bc,
                                                   float* __restrict__ Cc) {
  size_t id = (size_t)blockIdx.x * 256 + threadIdx.x;
  if (id >= (size_t)NROWS * CONVCH) return;
  int ch = (int)(id % CONVCH);
  size_t bt = id / CONVCH;
  int t = (int)(bt % SEQL);
  int b = (int)(bt / SEQL);
  const float* w = conv_w + ch * 4;
  float acc = conv_b[ch];
#pragma unroll
  for (int k = 0; k < 4; ++k) {
    int tt = t - 3 + k;
    if (tt >= 0) acc += w[k] * zx[((size_t)b * SEQL + tt) * DPROJ + DINNER + ch];
  }
  acc = silu_f(acc);
  size_t row = (size_t)b * SEQL + t;
  if (ch < DINNER)      xconv[row * DINNER + ch] = acc;
  else if (ch < DINNER + DSTATE) Bc[row * DSTATE + (ch - DINNER)] = acc;
  else                  Cc[row * DSTATE + (ch - DINNER - DSTATE)] = acc;
}

// ---------- per-chunk: cumsum + G=C*B^T, Y_diag, chunk states ----------
__global__ __launch_bounds__(256) void chunk_kernel(const float* __restrict__ zx,
                                                    const float* __restrict__ dt_bias,
                                                    const float* __restrict__ A_log,
                                                    const float* __restrict__ xconv,
                                                    const float* __restrict__ Bc,
                                                    const float* __restrict__ Cc,
                                                    float* __restrict__ y,
                                                    float* __restrict__ states,
                                                    float* __restrict__ Acs_g) {
  const int c = blockIdx.x, h = blockIdx.y, b = blockIdx.z;
  const int tid = threadIdx.x, lane = tid & 31, w = tid >> 5;
  const int mrow = lane & 15, hi = lane >> 4;
  __shared__ bf16_t sC[64][72], sB[64][72], sX[64][72], sM[64][72], sXT[64][72];
  __shared__ float scs[64], sdt[64];
  const size_t row0 = (size_t)b * SEQL + (size_t)c * 64;

  if (tid < 64) {
    float v = zx[(row0 + tid) * DPROJ + (DPROJ - NH) + h] + dt_bias[h];
    sdt[tid] = softplus_f(v);
  }
  __syncthreads();
  if (tid == 0) {
    float Ah = -__expf(A_log[h]);
    float a = 0.f;
    for (int t = 0; t < 64; ++t) { a += Ah * sdt[t]; scs[t] = a; }
  }
  __syncthreads();
  const float csT = scs[63];

  {
    int r = tid >> 2, c0 = (tid & 3) * 16;
#pragma unroll
    for (int e = 0; e < 16; ++e) {
      int col = c0 + e;
      sC[r][col] = f2bf(Cc[(row0 + r) * DSTATE + col]);
      sB[r][col] = f2bf(Bc[(row0 + r) * DSTATE + col]);
      float xv = xconv[(row0 + r) * DINNER + h * HD + col];
      float xd = xv * sdt[r];
      sX[r][col] = f2bf(xd);
      sXT[col][r] = f2bf(xd * __expf(csT - scs[r]));
    }
  }
  __syncthreads();

  // matmul 1: G = C (l x n) @ B^T (n x s); mask with L = exp(segsum)
#pragma unroll
  for (int i = 0; i < 2; ++i) {
    int ti = w * 2 + i, tl = ti >> 2, ts = ti & 3;
    v8f acc = zero8();
#pragma unroll
    for (int ks = 0; ks < 2; ++ks) {
      v16bf af, bfr;
#pragma unroll
      for (int e = 0; e < 16; ++e) af[e] = sC[tl * 16 + mrow][swzA(ks, hi, e)];
#pragma unroll
      for (int e = 0; e < 16; ++e) bfr[e] = sB[ts * 16 + mrow][swzB(ks, hi, e)];
      acc = __builtin_amdgcn_wmma_f32_16x16x32_bf16(false, af, false, bfr, (short)0, acc, false, false);
    }
#pragma unroll
    for (int r = 0; r < 8; ++r) {
      int l = tl * 16 + r + hi * 8, s = ts * 16 + mrow;
      float g = (l >= s) ? acc[r] * __expf(scs[l] - scs[s]) : 0.f;
      sM[l][s] = f2bf(g);
    }
  }
  __syncthreads();

  // matmul 2: Y_diag = M (l x s) @ Xdt (s x p)
#pragma unroll
  for (int i = 0; i < 2; ++i) {
    int ti = w * 2 + i, tl = ti >> 2, tp = ti & 3;
    v8f acc = zero8();
#pragma unroll
    for (int ks = 0; ks < 2; ++ks) {
      v16bf af, bfr;
#pragma unroll
      for (int e = 0; e < 16; ++e) af[e] = sM[tl * 16 + mrow][swzA(ks, hi, e)];
#pragma unroll
      for (int e = 0; e < 16; ++e) bfr[e] = sX[ks * 32 + hi * 16 + e][tp * 16 + mrow];
      acc = __builtin_amdgcn_wmma_f32_16x16x32_bf16(false, af, false, bfr, (short)0, acc, false, false);
    }
#pragma unroll
    for (int r = 0; r < 8; ++r) {
      int l = tl * 16 + r + hi * 8, p = tp * 16 + mrow;
      y[(row0 + l) * DINNER + h * HD + p] = acc[r];
    }
  }

  // matmul 3: S = (Xdt*decay)^T (p x t) @ B (t x n)
  const size_t sbase = ((size_t)(b * NC + c) * NH + h) * 4096;
#pragma unroll
  for (int i = 0; i < 2; ++i) {
    int ti = w * 2 + i, tp = ti >> 2, tn = ti & 3;
    v8f acc = zero8();
#pragma unroll
    for (int ks = 0; ks < 2; ++ks) {
      v16bf af, bfr;
#pragma unroll
      for (int e = 0; e < 16; ++e) af[e] = sXT[tp * 16 + mrow][swzA(ks, hi, e)];
#pragma unroll
      for (int e = 0; e < 16; ++e) bfr[e] = sB[ks * 32 + hi * 16 + e][tn * 16 + mrow];
      acc = __builtin_amdgcn_wmma_f32_16x16x32_bf16(false, af, false, bfr, (short)0, acc, false, false);
    }
#pragma unroll
    for (int r = 0; r < 8; ++r) {
      int p = tp * 16 + r + hi * 8, n = tn * 16 + mrow;
      states[sbase + (size_t)p * 64 + n] = acc[r];
    }
  }
  if (tid < 64)
    Acs_g[(((size_t)b * NH + h) * NC + c) * 64 + tid] = scs[tid];
}

// ---------- sequential inter-chunk state scan (exclusive prefix, in place) ----------
__global__ __launch_bounds__(256) void scan_kernel(float* __restrict__ states,
                                                   const float* __restrict__ Acs_g) {
  const int h = blockIdx.x, b = blockIdx.y;
  const int tid = threadIdx.x;
  float S[16];
#pragma unroll
  for (int i = 0; i < 16; ++i) S[i] = 0.f;
  for (int c = 0; c < NC; ++c) {
    float lam = __expf(Acs_g[(((size_t)b * NH + h) * NC + c) * 64 + 63]);
    size_t base = ((size_t)(b * NC + c) * NH + h) * 4096 + (size_t)tid * 16;
#pragma unroll
    for (int i = 0; i < 16; ++i) {
      float sc = states[base + i];
      float prev = S[i];
      states[base + i] = prev;           // exclusive prefix (state entering chunk c)
      S[i] = lam * prev + sc;
    }
  }
}

// ---------- Y_off = diag(exp(Acs)) * (C @ Sprev^T) + x*D, added into y ----------
__global__ __launch_bounds__(256) void yoff_kernel(const float* __restrict__ Cc,
                                                   const float* __restrict__ states,
                                                   const float* __restrict__ Acs_g,
                                                   const float* __restrict__ xconv,
                                                   const float* __restrict__ Dv,
                                                   float* __restrict__ y) {
  const int c = blockIdx.x, h = blockIdx.y, b = blockIdx.z;
  const int tid = threadIdx.x, lane = tid & 31, w = tid >> 5;
  const int mrow = lane & 15, hi = lane >> 4;
  __shared__ bf16_t sC[64][72], sS[64][72];
  __shared__ float scs[64];
  const size_t row0 = (size_t)b * SEQL + (size_t)c * 64;
  const size_t sbase = ((size_t)(b * NC + c) * NH + h) * 4096;

  if (tid < 64) scs[tid] = Acs_g[(((size_t)b * NH + h) * NC + c) * 64 + tid];
  {
    int r = tid >> 2, c0 = (tid & 3) * 16;
#pragma unroll
    for (int e = 0; e < 16; ++e) {
      int col = c0 + e;
      sC[r][col] = f2bf(Cc[(row0 + r) * DSTATE + col]);
      sS[r][col] = f2bf(states[sbase + (size_t)r * 64 + col]);  // sS[p][n]
    }
  }
  __syncthreads();
  const float Dh = Dv[h];
#pragma unroll
  for (int i = 0; i < 2; ++i) {
    int ti = w * 2 + i, tl = ti >> 2, tp = ti & 3;
    v8f acc = zero8();
#pragma unroll
    for (int ks = 0; ks < 2; ++ks) {
      v16bf af, bfr;
#pragma unroll
      for (int e = 0; e < 16; ++e) af[e] = sC[tl * 16 + mrow][swzA(ks, hi, e)];
#pragma unroll
      for (int e = 0; e < 16; ++e) bfr[e] = sS[tp * 16 + mrow][swzB(ks, hi, e)];
      acc = __builtin_amdgcn_wmma_f32_16x16x32_bf16(false, af, false, bfr, (short)0, acc, false, false);
    }
#pragma unroll
    for (int r = 0; r < 8; ++r) {
      int l = tl * 16 + r + hi * 8, p = tp * 16 + mrow;
      size_t gi = (row0 + l) * DINNER + h * HD + p;
      float xv = xconv[gi];
      y[gi] = y[gi] + __expf(scs[l]) * acc[r] + xv * Dh;
    }
  }
}

// ---------- gate with SiLU(z), RMSNorm, emit bf16 ----------
__global__ __launch_bounds__(256) void norm_kernel(const float* __restrict__ y,
                                                   const float* __restrict__ zx,
                                                   const float* __restrict__ norm_w,
                                                   bf16_t* __restrict__ ynorm) {
  const size_t row = blockIdx.x;
  const int tid = threadIdx.x;
  float vals[6];
  float sq = 0.f;
#pragma unroll
  for (int i = 0; i < 6; ++i) {
    int col = tid + i * 256;
    float z = zx[row * DPROJ + col];
    float v = y[row * DINNER + col] * silu_f(z);
    vals[i] = v;
    sq += v * v;
  }
  __shared__ float red[256];
  red[tid] = sq;
  __syncthreads();
  for (int s = 128; s > 0; s >>= 1) {
    if (tid < s) red[tid] += red[tid + s];
    __syncthreads();
  }
  float scale = rsqrtf(red[0] / (float)DINNER + 1e-5f);
#pragma unroll
  for (int i = 0; i < 6; ++i) {
    int col = tid + i * 256;
    ynorm[row * DINNER + col] = f2bf(vals[i] * scale * norm_w[col]);
  }
}

// ---------- host launch ----------
extern "C" void kernel_launch(void* const* d_in, const int* in_sizes, int n_in,
                              void* d_out, int out_size, void* d_ws, size_t ws_size,
                              hipStream_t stream) {
  const float* u       = (const float*)d_in[0];
  const float* W_in    = (const float*)d_in[1];
  const float* conv_w  = (const float*)d_in[2];
  const float* conv_b  = (const float*)d_in[3];
  const float* dt_bias = (const float*)d_in[4];
  const float* A_log   = (const float*)d_in[5];
  const float* Dv      = (const float*)d_in[6];
  const float* norm_w  = (const float*)d_in[7];
  const float* W_out   = (const float*)d_in[8];
  float* out = (float*)d_out;

  char* base = (char*)d_ws;
  size_t off = 0;
  auto take = [&](size_t bytes) { char* p = base + off; off = (off + bytes + 255) & ~(size_t)255; return p; };
  float*  zxbcdt  = (float*)take((size_t)NROWS * DPROJ * 4);
  float*  xconv   = (float*)take((size_t)NROWS * DINNER * 4);
  float*  Bc      = (float*)take((size_t)NROWS * DSTATE * 4);
  float*  Cc      = (float*)take((size_t)NROWS * DSTATE * 4);
  float*  ybuf    = (float*)take((size_t)NROWS * DINNER * 4);
  float*  states  = (float*)take((size_t)BATCH * NC * NH * 4096 * 4);
  float*  Acs     = (float*)take((size_t)BATCH * NH * NC * 64 * 4);
  bf16_t* u_bf    = (bf16_t*)take((size_t)NROWS * DMODEL * 2);
  bf16_t* win_bf  = (bf16_t*)take((size_t)DPROJ * DMODEL * 2);
  bf16_t* yn_bf   = (bf16_t*)take((size_t)NROWS * DINNER * 2);
  bf16_t* wout_bf = (bf16_t*)take((size_t)DMODEL * DINNER * 2);

  // 1) convert operands to bf16
  {
    size_t n = (size_t)NROWS * DMODEL;
    convert_kernel<<<(unsigned)((n + 255) / 256), 256, 0, stream>>>(u, u_bf, n);
    n = (size_t)DPROJ * DMODEL;
    convert_kernel<<<(unsigned)((n + 255) / 256), 256, 0, stream>>>(W_in, win_bf, n);
    n = (size_t)DMODEL * DINNER;
    convert_kernel<<<(unsigned)((n + 255) / 256), 256, 0, stream>>>(W_out, wout_bf, n);
  }
  // 2) in-projection GEMM: zxbcdt = u @ W_in^T   (TDM-staged, double-buffered)
  gemm_bf16<<<dim3((DPROJ + 127) / 128, NROWS / 128), 256, 0, stream>>>(
      u_bf, win_bf, zxbcdt, NROWS, DPROJ, DMODEL, DPROJ);
  // 3) conv + SiLU + split
  {
    size_t n = (size_t)NROWS * CONVCH;
    conv_kernel<<<(unsigned)((n + 255) / 256), 256, 0, stream>>>(zxbcdt, conv_w, conv_b, xconv, Bc, Cc);
  }
  // 4) per-chunk WMMA stage
  chunk_kernel<<<dim3(NC, NH, BATCH), 256, 0, stream>>>(
      zxbcdt, dt_bias, A_log, xconv, Bc, Cc, ybuf, states, Acs);
  // 5) inter-chunk state scan
  scan_kernel<<<dim3(NH, BATCH), 256, 0, stream>>>(states, Acs);
  // 6) Y_off + skip term
  yoff_kernel<<<dim3(NC, NH, BATCH), 256, 0, stream>>>(Cc, states, Acs, xconv, Dv, ybuf);
  // 7) gate + RMSNorm -> bf16
  norm_kernel<<<NROWS, 256, 0, stream>>>(ybuf, zxbcdt, norm_w, yn_bf);
  // 8) out-projection GEMM
  gemm_bf16<<<dim3(DMODEL / 128, NROWS / 128), 256, 0, stream>>>(
      yn_bf, wout_bf, out, NROWS, DMODEL, DINNER, DMODEL);
}
